// Decoder_40424232190400
// MI455X (gfx1250) — matrix operand order repaired and checked by reading, not compiled, not run on hardware
//
#include <hip/hip_runtime.h>

// ---------------- dims ----------------
#define BQ    32
#define TENC  200
#define STEPS 200
#define NMEL  80
#define NW    16          // waves per block (wave32)
#define BLK   512

typedef __attribute__((ext_vector_type(16))) __bf16 v16bf;
typedef __attribute__((ext_vector_type(8)))  float  v8f;

union FragU { v16bf v; unsigned int u[8]; };

// ---------------- workspace layout (element units in one buffer) ----------------
// bf16-pair fragment regions (u32 units)
#define CNT_AL  (20*64*256)
#define CNT_DL  (24*64*256)
#define CNT_P1  (3*16*256)
#define CNT_P2  (8*8*256)
#define CNT_AW  (8*8*256)
#define CNT_AV  (4*11*256)
#define CNT_EW  (1*8*256)
#define CNT_AC  (16*10*256)
#define CNT_MEM (32*7*16*256)

#define OFF_ALFRAG 0
#define OFF_DLFRAG (OFF_ALFRAG + CNT_AL)
#define OFF_P1FRAG (OFF_DLFRAG + CNT_DL)
#define OFF_P2FRAG (OFF_P1FRAG + CNT_P1)
#define OFF_AWFRAG (OFF_P2FRAG + CNT_P2)
#define OFF_AVFRAG (OFF_AWFRAG + CNT_AW)
#define OFF_EWFRAG (OFF_AVFRAG + CNT_AV)
#define OFF_ACFRAG (OFF_EWFRAG + CNT_EW)
#define OFF_MEMFRAG (OFF_ACFRAG + CNT_AC)
#define FRAG_TOTAL  (OFF_MEMFRAG + CNT_MEM)

// fp32 regions (float units, same buffer)
#define OFF_ALBIAS (FRAG_TOTAL)
#define OFF_DLBIAS (OFF_ALBIAS + 1024)
#define OFF_H1   (OFF_DLBIAS + 1024)
#define OFF_C1   (OFF_H1 + 8192)
#define OFF_H2   (OFF_C1 + 8192)
#define OFF_C2   (OFF_H2 + 8192)
#define OFF_CTX  (OFF_C2 + 8192)
#define OFF_ALGN (OFF_CTX + 8192)          // 32 x 200
#define OFF_PREH (OFF_ALGN + 6400)         // 32 x 256
#define OFF_PRE  (OFF_PREH + 8192)         // 32 x 128
#define OFF_GATE (OFF_PRE + 4096)          // 32 x 1024
#define OFF_GTMP (OFF_GATE + 32768)        // 32 x 128
#define OFF_G    (OFF_GTMP + 4096)         // 32 x 176
#define OFF_PW   (OFF_G + 5632)            // 32 x 200  log-prior
#define OFF_EACC (OFF_PW + 6400)           // 32 x 208  energies
#define OFF_EIN  (OFF_EACC + 6656)         // 32 x 208 x 16 conv features
#define WS_ELEMS (OFF_EIN + 106496)

#define OUT_ALIGN_OFF (32*80*400)

// ---------------- helpers ----------------
__device__ __forceinline__ unsigned int pack_bf16(float lo, float hi) {
  unsigned int a = __builtin_bit_cast(unsigned int, lo);
  unsigned int b = __builtin_bit_cast(unsigned int, hi);
  a = (a + 0x7FFFu + ((a >> 16) & 1u)) >> 16;     // RNE f32->bf16
  b = (b + 0x7FFFu + ((b >> 16) & 1u)) >> 16;
  return (a & 0xFFFFu) | (b << 16);
}

__device__ __forceinline__ float sigm(float x) { return 1.0f / (1.0f + __expf(-x)); }

__device__ __forceinline__ v8f zero8() {
  v8f a;
#pragma unroll
  for (int i = 0; i < 8; ++i) a[i] = 0.0f;
  return a;
}

// A-fragment (16x32 bf16): lane = {row M = lane&15, half = lane>>4};
// VGPR r holds K pair K0 = 16*(r>>2) + 2*(r&3) + 8*half  (pair idx = K0/2)
__device__ __forceinline__ void load_afrag(FragU& a, const unsigned int* __restrict__ row,
                                           int kt, int half) {
  const int base = kt * 16 + 4 * half;
#pragma unroll
  for (int r = 0; r < 8; ++r) a.u[r] = row[base + 8 * (r >> 2) + (r & 3)];
}

// B-fragment (32x16 bf16): lane = {col N = lane&15, half}; VGPR r holds K = 16*(r>>2) + 2*(r&3) + 8*half
// Stored fragment-major: (((kt*NT)+nt)*32 + lane)*8 + r   -> coalesced b128 loads
__device__ __forceinline__ void load_bfrag(FragU& b, const unsigned int* __restrict__ wf,
                                           int NT, int kt, int nt, int lane) {
  const unsigned int* p = wf + ((((long)kt * NT) + nt) * 32 + lane) * 8;
  __builtin_prefetch(p + 2048, 0, 1);   // pull next tiles toward L0 (global_prefetch_b8)
#pragma unroll
  for (int r = 0; r < 8; ++r) b.u[r] = p[r];
}

__device__ __forceinline__ v8f wmma_bf16(const FragU& a, const FragU& b, v8f acc) {
  return __builtin_amdgcn_wmma_f32_16x16x32_bf16(false, a.v, false, b.v, (short)0, acc,
                                                 false, false);
}

// ---------------- prep: weights/memory -> bf16 B-fragments ----------------
__global__ void prep_kernel(const float* __restrict__ pre_W1, const float* __restrict__ pre_W2,
                            const float* __restrict__ al_Wih, const float* __restrict__ al_Whh,
                            const float* __restrict__ al_bih, const float* __restrict__ al_bhh,
                            const float* __restrict__ att_W,  const float* __restrict__ att_V,
                            const float* __restrict__ att_U,  const float* __restrict__ att_T,
                            const float* __restrict__ dl_Wih, const float* __restrict__ dl_Whh,
                            const float* __restrict__ dl_bih, const float* __restrict__ dl_bhh,
                            const float* __restrict__ ac_W,   const float* __restrict__ memory,
                            unsigned int* __restrict__ wsU) {
  float* wsF = (float*)wsU;
  const int total = FRAG_TOTAL + 2048;
  for (int id = blockIdx.x * blockDim.x + threadIdx.x; id < total;
       id += gridDim.x * blockDim.x) {
    if (id >= FRAG_TOTAL) {
      int j = id - FRAG_TOTAL;
      if (j < 1024) wsF[OFF_ALBIAS + j] = al_bih[j] + al_bhh[j];
      else          wsF[OFF_DLBIAS + (j - 1024)] = dl_bih[j - 1024] + dl_bhh[j - 1024];
      continue;
    }
    int x = id;
    auto emit = [&](int off, int NT, auto W) {
      int kt = x / (NT * 256);
      int rem = x - kt * NT * 256;
      int nt = rem >> 8, li = rem & 255, ln = li >> 3, r = li & 7;
      int K0 = kt * 32 + 16 * (r >> 2) + 2 * (r & 3) + 8 * (ln >> 4);
      int n = nt * 16 + (ln & 15);
      wsU[off + x] = pack_bf16(W(K0, n), W(K0 + 1, n));
    };
    if (x < CNT_AL) {                                // [ctx|pre|h1](640) -> 1024 gates
      emit(OFF_ALFRAG, 64, [&](int k, int n) {
        return (k < 384) ? al_Wih[n * 384 + k] : al_Whh[n * 256 + (k - 384)]; });
    } else if ((x -= CNT_AL) < CNT_DL) {             // [h1|ctx|h2](768) -> 1024 gates
      emit(OFF_DLFRAG, 64, [&](int k, int n) {
        return (k < 512) ? dl_Wih[n * 512 + k] : dl_Whh[n * 256 + (k - 512)]; });
    } else if ((x -= CNT_DL) < CNT_P1) {             // mels(80->96) -> 256
      emit(OFF_P1FRAG, 16, [&](int k, int n) {
        return (k < 80) ? pre_W1[n * 80 + k] : 0.0f; });
    } else if ((x -= CNT_P1) < CNT_P2) {             // 256 -> 128
      emit(OFF_P2FRAG, 8, [&](int k, int n) { return pre_W2[n * 256 + k]; });
    } else if ((x -= CNT_P2) < CNT_AW) {             // 256 -> 128
      emit(OFF_AWFRAG, 8, [&](int k, int n) { return att_W[n * 256 + k]; });
    } else if ((x -= CNT_AW) < CNT_AV) {             // 128 -> 168(176)
      emit(OFF_AVFRAG, 11, [&](int k, int n) {
        return (n < 168) ? att_V[n * 128 + k] : 0.0f; });
    } else if ((x -= CNT_AV) < CNT_EW) {             // [f(8)|g(8)](16->32) -> 128
      emit(OFF_EWFRAG, 8, [&](int k, int n) {
        if (k < 8)  return att_U[n * 8 + k];
        if (k < 16) return att_T[n * 8 + (k - 8)];
        return 0.0f; });
    } else if ((x -= CNT_EW) < CNT_AC) {             // [h2|ctx](512) -> 160
      emit(OFF_ACFRAG, 10, [&](int k, int n) { return ac_W[n * 512 + k]; });
    } else {                                         // memory[b]: 224(K=t) x 256(N=d)
      x -= CNT_AC;
      int b = x / (7 * 16 * 256);
      int loc = x - b * 7 * 16 * 256;
      int kt = loc / (16 * 256);
      int rem = loc - kt * 16 * 256;
      int nt = rem >> 8, li = rem & 255, ln = li >> 3, r = li & 7;
      int t0 = kt * 32 + 16 * (r >> 2) + 2 * (r & 3) + 8 * (ln >> 4);
      int d = nt * 16 + (ln & 15);
      float v0 = (t0 < TENC) ? memory[(b * TENC + t0) * 256 + d] : 0.0f;
      float v1 = (t0 + 1 < TENC) ? memory[(b * TENC + t0 + 1) * 256 + d] : 0.0f;
      wsU[OFF_MEMFRAG + x] = pack_bf16(v0, v1);
    }
  }
}

// ---------------- persistent single-WGP decoder ----------------
__global__ __launch_bounds__(BLK) void decoder_kernel(
    const float* __restrict__ mels,   const float* __restrict__ pre_b1,
    const float* __restrict__ pre_b2, const float* __restrict__ att_Wb,
    const float* __restrict__ att_F,  const float* __restrict__ att_Tb,
    const float* __restrict__ att_v,  const float* __restrict__ prior,
    const float* __restrict__ ac_b,   unsigned int* __restrict__ wsU,
    float* __restrict__ out) {
  float* wsF = (float*)wsU;
  const int tid = threadIdx.x, lane = tid & 31, wave = tid >> 5;
  const int m16 = lane & 15, half = lane >> 4;

  __shared__ unsigned int actp[32][384];   // per-row bf16 pairs (K up to 768)

  const unsigned int* alfrag = wsU + OFF_ALFRAG;
  const unsigned int* dlfrag = wsU + OFF_DLFRAG;
  const unsigned int* p1frag = wsU + OFF_P1FRAG;
  const unsigned int* p2frag = wsU + OFF_P2FRAG;
  const unsigned int* awfrag = wsU + OFF_AWFRAG;
  const unsigned int* avfrag = wsU + OFF_AVFRAG;
  const unsigned int* ewfrag = wsU + OFF_EWFRAG;
  const unsigned int* acfrag = wsU + OFF_ACFRAG;
  const unsigned int* memfrag = wsU + OFF_MEMFRAG;
  const float* albias = wsF + OFF_ALBIAS;
  const float* dlbias = wsF + OFF_DLBIAS;
  float* h1 = wsF + OFF_H1;  float* c1 = wsF + OFF_C1;
  float* h2 = wsF + OFF_H2;  float* c2 = wsF + OFF_C2;
  float* ctx = wsF + OFF_CTX;
  float* alignA = wsF + OFF_ALGN;
  float* preh = wsF + OFF_PREH; float* pre = wsF + OFF_PRE;
  float* gates = wsF + OFF_GATE;
  float* Gtmp = wsF + OFF_GTMP; float* Gw = wsF + OFF_G;
  float* pw = wsF + OFF_PW;     float* eacc = wsF + OFF_EACC;
  float* ein = wsF + OFF_EIN;

  // stage fp32 source into LDS bf16-pair A layout
  auto stage = [&](int Kp, auto f) {
    const int pairs = Kp >> 1;
    for (int i = tid; i < 32 * pairs; i += BLK) {
      int b = i / pairs, p = i - b * pairs;
      actp[b][p] = pack_bf16(f(b, 2 * p), f(b, 2 * p + 1));
    }
  };
  // generic WMMA GEMM: act(LDS) x wfrags -> post(row,col,val)
  auto run_gemm = [&](const unsigned int* wf, int MT, int NT, int KT, auto post) {
    for (int t = wave; t < MT * NT; t += NW) {
      int mt = t / NT, nt = t - mt * NT;
      v8f acc = zero8();
      const unsigned int* arow = actp[mt * 16 + m16];
      for (int kt = 0; kt < KT; ++kt) {
        FragU a, bfr;
        load_afrag(a, arow, kt, half);
        load_bfrag(bfr, wf, NT, kt, nt, lane);
        acc = wmma_bf16(a, bfr, acc);
      }
#pragma unroll
      for (int r = 0; r < 8; ++r) post(mt * 16 + r + 8 * half, nt * 16 + m16, acc[r]);
    }
  };

  // ---- init state ----
  for (int i = tid; i < 8192; i += BLK) { h1[i]=0; c1[i]=0; h2[i]=0; c2[i]=0; ctx[i]=0; }
  for (int i = tid; i < BQ * TENC; i += BLK) alignA[i] = ((i % TENC) == 0) ? 1.0f : 0.0f;
  __syncthreads();

  for (int s = 0; s < STEPS; ++s) {
    // --- prenet: x_t = mels[:,:,2s-1] (zeros at s=0) ---
    stage(96, [&](int b, int k) {
      return (s > 0 && k < NMEL) ? mels[b * 32000 + k * 400 + (2 * s - 1)] : 0.0f; });
    __syncthreads();
    run_gemm(p1frag, 2, 16, 3, [&](int row, int col, float v) {
      preh[row * 256 + col] = fmaxf(v + pre_b1[col], 0.0f); });
    __syncthreads();
    stage(256, [&](int b, int k) { return preh[b * 256 + k]; });
    __syncthreads();
    run_gemm(p2frag, 2, 8, 8, [&](int row, int col, float v) {
      pre[row * 128 + col] = fmaxf(v + pre_b2[col], 0.0f); });
    __syncthreads();

    // --- attention LSTM gates: act = [ctx | pre | h1] ---
    stage(640, [&](int b, int k) {
      if (k < 256) return ctx[b * 256 + k];
      if (k < 384) return pre[b * 128 + (k - 256)];
      return h1[b * 256 + (k - 384)]; });
    __syncthreads();
    run_gemm(alfrag, 2, 64, 20, [&](int row, int col, float v) {
      gates[row * 1024 + col] = v + albias[col]; });
    __syncthreads();
    for (int i = tid; i < 8192; i += BLK) {            // LSTM1 elementwise
      int b = i >> 8, n = i & 255;
      const float* g = gates + b * 1024;
      float c = sigm(g[256 + n]) * c1[i] + sigm(g[n]) * tanhf(g[512 + n]);
      c1[i] = c;  h1[i] = sigm(g[768 + n]) * tanhf(c);
    }
    __syncthreads();

    // --- DCA: prior conv (VALU) + stage h1 for G path ---
    stage(256, [&](int b, int k) { return h1[b * 256 + k]; });
    for (int i = tid; i < BQ * TENC; i += BLK) {
      int b = i / TENC, t = i - b * TENC;
      float p = 0.0f;
#pragma unroll
      for (int m = 0; m < 11; ++m)
        if (t - m >= 0) p += prior[m] * alignA[b * TENC + t - m];
      pw[i] = __logf(fmaxf(p, 1e-6f));
    }
    __syncthreads();
    run_gemm(awfrag, 2, 8, 8, [&](int row, int col, float v) {
      Gtmp[row * 128 + col] = tanhf(v + att_Wb[col]); });
    __syncthreads();
    stage(128, [&](int b, int k) { return Gtmp[b * 128 + k]; });
    __syncthreads();
    run_gemm(avfrag, 2, 11, 4, [&](int row, int col, float v) { Gw[row * 176 + col] = v; });
    __syncthreads();

    // --- static + dynamic conv as Toeplitz WMMA: out ein[b][t][0:8]=static, [8:16]=dyn ---
    for (int b = wave; b < BQ; b += NW) {
      FragU bfr;                                       // B: taps(21->32) x 16 filters
#pragma unroll
      for (int r = 0; r < 8; ++r) {
        int k = 16 * (r >> 2) + 2 * (r & 3) + 8 * half;
        float v0 = 0.0f, v1 = 0.0f;
        if (k < 21)
          v0 = (m16 < 8) ? att_F[m16 * 21 + k] : Gw[b * 176 + (m16 - 8) * 21 + k];
        if (k + 1 < 21)
          v1 = (m16 < 8) ? att_F[m16 * 21 + k + 1] : Gw[b * 176 + (m16 - 8) * 21 + k + 1];
        bfr.u[r] = pack_bf16(v0, v1);
      }
      for (int mt = 0; mt < 13; ++mt) {
        FragU a;
        int t = mt * 16 + m16;
#pragma unroll
        for (int r = 0; r < 8; ++r) {                  // Toeplitz row of align
          int K0 = 16 * (r >> 2) + 2 * (r & 3) + 8 * half;
          int i0 = t + K0 - 10, i1 = i0 + 1;
          float v0 = (K0 < 21 && i0 >= 0 && i0 < TENC) ? alignA[b * TENC + i0] : 0.0f;
          float v1 = (K0 + 1 < 21 && i1 >= 0 && i1 < TENC) ? alignA[b * TENC + i1] : 0.0f;
          a.u[r] = pack_bf16(v0, v1);
        }
        v8f acc = wmma_bf16(a, bfr, zero8());
#pragma unroll
        for (int r = 0; r < 8; ++r)
          ein[(b * 208 + mt * 16 + r + 8 * half) * 16 + m16] = acc[r];
      }
    }
    __syncthreads();

    // --- energy GEMM [6656x16]x[16(->32)x128] + tanh + dot(att_v), deterministic ---
    for (int mtg = wave; mtg < 416; mtg += NW) {
      int b = mtg / 13, mtl = mtg - b * 13;
      FragU a;
      const float* er = ein + (b * 208 + mtl * 16 + m16) * 16;
#pragma unroll
      for (int r = 0; r < 8; ++r) {
        int K0 = 16 * (r >> 2) + 2 * (r & 3) + 8 * half;
        a.u[r] = (K0 < 16) ? pack_bf16(er[K0], er[K0 + 1]) : 0u;
      }
      float esum[8];
#pragma unroll
      for (int r = 0; r < 8; ++r) esum[r] = 0.0f;
      for (int nt = 0; nt < 8; ++nt) {
        FragU bfr;
        load_bfrag(bfr, ewfrag, 8, 0, nt, lane);
        v8f acc = wmma_bf16(a, bfr, zero8());
        int n = nt * 16 + m16;
        float tb = att_Tb[n], vv = att_v[n];
#pragma unroll
        for (int r = 0; r < 8; ++r) {
          float e = tanhf(acc[r] + tb) * vv;
          e += __shfl_xor(e, 8, 16);
          e += __shfl_xor(e, 4, 16);
          e += __shfl_xor(e, 2, 16);
          e += __shfl_xor(e, 1, 16);
          esum[r] += e;
        }
      }
#pragma unroll
      for (int r = 0; r < 8; ++r)
        if (m16 == 0) eacc[b * 208 + mtl * 16 + r + 8 * half] = esum[r];
    }
    __syncthreads();

    // --- softmax(e + log-prior) per batch row; emit alignment output ---
    for (int b = wave; b < BQ; b += NW) {
      float mx = -1e30f;
      for (int t = lane; t < TENC; t += 32)
        mx = fmaxf(mx, eacc[b * 208 + t] + pw[b * TENC + t]);
#pragma unroll
      for (int o = 16; o > 0; o >>= 1) mx = fmaxf(mx, __shfl_xor(mx, o, 32));
      float sum = 0.0f;
      for (int t = lane; t < TENC; t += 32)
        sum += __expf(eacc[b * 208 + t] + pw[b * TENC + t] - mx);
#pragma unroll
      for (int o = 16; o > 0; o >>= 1) sum += __shfl_xor(sum, o, 32);
      float inv = 1.0f / sum;
      for (int t = lane; t < TENC; t += 32) {
        float av = __expf(eacc[b * 208 + t] + pw[b * TENC + t] - mx) * inv;
        alignA[b * TENC + t] = av;
        out[OUT_ALIGN_OFF + (b * TENC + t) * STEPS + s] = av;
      }
    }
    __syncthreads();

    // --- ctx = align @ memory[b]  (M=1-in-16 WMMA vs pre-swizzled memory frags) ---
    for (int j = wave; j < 512; j += NW) {
      int b = j >> 4, nt = j & 15;
      v8f acc = zero8();
      for (int kt = 0; kt < 7; ++kt) {
        FragU a, bfr;
#pragma unroll
        for (int r = 0; r < 8; ++r) {
          unsigned int pv = 0u;
          if (m16 == 0) {                              // only A row 0 is live
            int K0 = kt * 32 + 16 * (r >> 2) + 2 * (r & 3) + 8 * half;
            float v0 = (K0 < TENC) ? alignA[b * TENC + K0] : 0.0f;
            float v1 = (K0 + 1 < TENC) ? alignA[b * TENC + K0 + 1] : 0.0f;
            pv = pack_bf16(v0, v1);
          }
          a.u[r] = pv;
        }
        load_bfrag(bfr, memfrag + b * (7 * 16 * 256), 16, kt, nt, lane);
        acc = wmma_bf16(a, bfr, acc);
      }
      if (half == 0) ctx[b * 256 + nt * 16 + m16] = acc[0];   // C row 0
    }
    __syncthreads();

    // --- decoder LSTM gates: act = [h1 | ctx | h2] ---
    stage(768, [&](int b, int k) {
      if (k < 256) return h1[b * 256 + k];
      if (k < 512) return ctx[b * 256 + (k - 256)];
      return h2[b * 256 + (k - 512)]; });
    __syncthreads();
    run_gemm(dlfrag, 2, 64, 24, [&](int row, int col, float v) {
      gates[row * 1024 + col] = v + dlbias[col]; });
    __syncthreads();
    for (int i = tid; i < 8192; i += BLK) {            // LSTM2 elementwise
      int b = i >> 8, n = i & 255;
      const float* g = gates + b * 1024;
      float c = sigm(g[256 + n]) * c2[i] + sigm(g[n]) * tanhf(g[512 + n]);
      c2[i] = c;  h2[i] = sigm(g[768 + n]) * tanhf(c);
    }
    __syncthreads();

    // --- acoustic projection fused in-step: [h2 | ctx] -> 160, scatter to mels out ---
    stage(512, [&](int b, int k) {
      return (k < 256) ? h2[b * 256 + k] : ctx[b * 256 + (k - 256)]; });
    __syncthreads();
    run_gemm(acfrag, 2, 10, 16, [&](int row, int col, float v) {
      int nm = col >> 1, ri = col & 1;                 // out[b, n*2+r] -> mels[b][n][2s+r]
      out[(row * NMEL + nm) * 400 + 2 * s + ri] = v + ac_b[col]; });
    __syncthreads();
  }
}

// ---------------- launch ----------------
extern "C" void kernel_launch(void* const* d_in, const int* in_sizes, int n_in,
                              void* d_out, int out_size, void* d_ws, size_t ws_size,
                              hipStream_t stream) {
  (void)in_sizes; (void)n_in; (void)out_size; (void)ws_size;
  const float* mels   = (const float*)d_in[0];
  const float* memory = (const float*)d_in[1];
  const float* pre_W1 = (const float*)d_in[2];
  const float* pre_b1 = (const float*)d_in[3];
  const float* pre_W2 = (const float*)d_in[4];
  const float* pre_b2 = (const float*)d_in[5];
  const float* al_Wih = (const float*)d_in[6];
  const float* al_Whh = (const float*)d_in[7];
  const float* al_bih = (const float*)d_in[8];
  const float* al_bhh = (const float*)d_in[9];
  const float* att_W  = (const float*)d_in[10];
  const float* att_Wb = (const float*)d_in[11];
  const float* att_V  = (const float*)d_in[12];
  const float* att_F  = (const float*)d_in[13];
  const float* att_U  = (const float*)d_in[14];
  const float* att_T  = (const float*)d_in[15];
  const float* att_Tb = (const float*)d_in[16];
  const float* att_v  = (const float*)d_in[17];
  const float* prior  = (const float*)d_in[18];
  const float* dl_Wih = (const float*)d_in[19];
  const float* dl_Whh = (const float*)d_in[20];
  const float* dl_bih = (const float*)d_in[21];
  const float* dl_bhh = (const float*)d_in[22];
  const float* ac_W   = (const float*)d_in[23];
  const float* ac_b   = (const float*)d_in[24];
  unsigned int* wsU = (unsigned int*)d_ws;
  float* out = (float*)d_out;

  prep_kernel<<<2048, 256, 0, stream>>>(pre_W1, pre_W2, al_Wih, al_Whh, al_bih, al_bhh,
                                        att_W, att_V, att_U, att_T, dl_Wih, dl_Whh,
                                        dl_bih, dl_bhh, ac_W, memory, wsU);
  decoder_kernel<<<1, BLK, 0, stream>>>(mels, pre_b1, pre_b2, att_Wb, att_F, att_Tb,
                                        att_v, prior, ac_b, wsU, out);
}
